// StackedLSTMEncoder_21114059227672
// MI455X (gfx1250) — compile-verified, matrix operand order checked by reading, not run on hardware
//
#include <hip/hip_runtime.h>

#define VOCAB 32000
#define EMBED 512
#define HIDDEN 1024
#define BATCH 64
#define SEQ 256

typedef __attribute__((ext_vector_type(16))) __bf16 v16bf;
typedef __attribute__((ext_vector_type(8)))  float  v8f;
typedef __attribute__((ext_vector_type(4)))  unsigned int v4u;

union Frag { v4u u[2]; v16bf bf; };

// ---- workspace layout (bytes) ----
constexpr size_t WS_XBF  = 256;                                   // ctrl counters at 0
constexpr size_t WS_H0BF = WS_XBF  + 2ull * BATCH * EMBED  * 2;   // 2 x [64][512]  bf16
constexpr size_t WS_H1BF = WS_H0BF + 2ull * BATCH * HIDDEN * 2;   // 2 x [64][1024] bf16
// ---- d_out layout (floats) ----
constexpr size_t OUT_LC  = (size_t)BATCH * 2 * HIDDEN;            // last_cell after last_hidden
constexpr size_t OUT_SEQ = 2 * OUT_LC;                            // outputs [64][256][1024]

constexpr int NBLOCKS = 128;                                      // 64 layer0 + 64 layer1 groups
constexpr size_t SMEM_BYTES = 64ull * 2048 * 2 /*weights*/ + 64 * 64 * 4 /*gates*/ + 64 * 4 /*bias*/;

__device__ __forceinline__ unsigned short f2bf(float f) {
  unsigned u = __float_as_uint(f);
  unsigned r = u + 0x7FFFu + ((u >> 16) & 1u);   // round-to-nearest-even
  return (unsigned short)(r >> 16);
}
__device__ __forceinline__ float sigm(float x) { return 1.0f / (1.0f + __expf(-x)); }

// ---------------- init: zero barrier state, seed f32 state + bf16 mirrors, gather x(0) ----------------
__global__ void lstm_init(const int* __restrict__ tokens, const float* __restrict__ hidden,
                          const float* __restrict__ cell, const float* __restrict__ embed_w,
                          float* __restrict__ out, unsigned char* __restrict__ ws) {
  unsigned* ctrl = (unsigned*)ws;
  unsigned short* xbf0  = (unsigned short*)(ws + WS_XBF);                       // parity 0
  unsigned short* h0bf1 = (unsigned short*)(ws + WS_H0BF) + BATCH * HIDDEN;     // parity 1
  unsigned short* h1bf1 = (unsigned short*)(ws + WS_H1BF) + BATCH * HIDDEN;     // parity 1
  int tid = blockIdx.x * blockDim.x + threadIdx.x;
  int nth = gridDim.x * blockDim.x;
  if (tid == 0) { ctrl[0] = 0u; ctrl[1] = 0u; }
  for (int i = tid; i < BATCH * HIDDEN; i += nth) {
    int b = i / HIDDEN, j = i % HIDDEN;
    float h0 = hidden[(size_t)b * 2 * HIDDEN + j];
    float h1 = hidden[(size_t)b * 2 * HIDDEN + HIDDEN + j];
    h0bf1[i] = f2bf(h0);
    h1bf1[i] = f2bf(h1);
    out[(size_t)b * 2 * HIDDEN + j]                   = h0;
    out[(size_t)b * 2 * HIDDEN + HIDDEN + j]          = h1;
    out[OUT_LC + (size_t)b * 2 * HIDDEN + j]          = cell[(size_t)b * 2 * HIDDEN + j];
    out[OUT_LC + (size_t)b * 2 * HIDDEN + HIDDEN + j] = cell[(size_t)b * 2 * HIDDEN + HIDDEN + j];
  }
  for (int i = tid; i < BATCH * EMBED; i += nth) {
    int b = i / EMBED, e = i % EMBED;
    int tok = tokens[(size_t)b * SEQ + 0];
    xbf0[i] = f2bf(embed_w[(size_t)tok * EMBED + e]);
  }
}

// ---------------- persistent pipelined LSTM ----------------
__global__ void __launch_bounds__(256, 1)
lstm_main(const int* __restrict__ tokens, const float* __restrict__ embed_w,
          const float* __restrict__ W_ih0, const float* __restrict__ W_hh0, const float* __restrict__ b0,
          const float* __restrict__ W_ih1, const float* __restrict__ W_hh1, const float* __restrict__ b1,
          float* __restrict__ out, unsigned char* __restrict__ ws) {
  extern __shared__ unsigned char smem[];
  const int layer = blockIdx.x >> 6;      // 0: layer0 groups, 1: layer1 groups
  const int g     = blockIdx.x & 63;      // hidden-unit slice: units [g*16, g*16+16)
  const int tid   = threadIdx.x;
  const int lane  = tid & 31, wv = tid >> 5;
  const int lm    = lane & 15, kh = lane >> 4;

  const int Ktot = layer ? 2048 : 1536;   // concat K of [x|h] resp. [h0|h1]
  unsigned short* ldsW = (unsigned short*)smem;                 // [64 rows][Ktot] bf16, persistent
  float* ldsG = (float*)(smem + (size_t)Ktot * 64 * 2);         // [64 batch][64 gates] f32 scratch
  float* ldsB = ldsG + 64 * 64;                                 // [64] bias

  unsigned* ctrl = (unsigned*)ws;
  unsigned short* xbf  = (unsigned short*)(ws + WS_XBF);
  unsigned short* h0bf = (unsigned short*)(ws + WS_H0BF);
  unsigned short* h1bf = (unsigned short*)(ws + WS_H1BF);
  float* outLH  = out;
  float* outLC  = out + OUT_LC;
  float* outSeq = out + OUT_SEQ;

  // --- one-time: stage this group's weight slice f32->bf16 into LDS (rows lr = q*16+i <-> gate row q*H + g*16+i)
  {
    const float* Wih  = layer ? W_ih1 : W_ih0;
    const float* Whh  = layer ? W_hh1 : W_hh0;
    const float* bias = layer ? b1 : b0;
    const int Kih = layer ? HIDDEN : EMBED;
    for (int idx = tid; idx < 64 * Ktot; idx += 256) {
      int lr = idx / Ktot, k = idx - lr * Ktot;
      int grow = (lr >> 4) * HIDDEN + g * 16 + (lr & 15);
      float v = (k < Kih) ? Wih[(size_t)grow * Kih + k]
                          : Whh[(size_t)grow * HIDDEN + (k - Kih)];
      ldsW[idx] = f2bf(v);
    }
    for (int lr = tid; lr < 64; lr += 256)
      ldsB[lr] = bias[(lr >> 4) * HIDDEN + g * 16 + (lr & 15)];
  }
  __syncthreads();

  const int mi  = wv >> 1;          // batch tile 0..3 (16 rows each)
  const int nt0 = (wv & 1) * 2;     // this wave owns gate tiles nt0, nt0+1

  for (int p = 0; p <= SEQ; ++p) {
    const bool active = layer ? (p >= 1) : (p < SEQ);
    const int s = layer ? (p - 1) : p;    // LSTM step this group computes in this phase
    if (active) {
      // activation segments (bf16, double-buffered in ws)
      const unsigned short *actA, *actB; int KA, KB;
      if (layer == 0) {
        actA = xbf  + (size_t)(p & 1)       * BATCH * EMBED;  KA = EMBED;   // x(p)
        actB = h0bf + (size_t)((p + 1) & 1) * BATCH * HIDDEN; KB = HIDDEN;  // h0(p-1)
      } else {
        actA = h0bf + (size_t)(s & 1)       * BATCH * HIDDEN; KA = HIDDEN;  // h0(s)
        actB = h1bf + (size_t)((s + 1) & 1) * BATCH * HIDDEN; KB = HIDDEN;  // h1(s-1)
      }
      v8f acc0 = {}, acc1 = {};
      int wkoff = 0;
      for (int seg = 0; seg < 2; ++seg) {
        const unsigned short* act = seg ? actB : actA;
        const int K = seg ? KB : KA;
        for (int kt = 0; kt < K / 32; ++kt) {
          Frag a, f0, f1;
          const unsigned short* ap = act + (size_t)(mi * 16 + lm) * K + kt * 32 + kh * 16;
          a.u[0] = *(const v4u*)ap;                 // global_load_b128 (L2-hot activations)
          a.u[1] = *(const v4u*)(ap + 8);
          const unsigned short* bp0 = ldsW + (size_t)(nt0 * 16 + lm) * Ktot + wkoff + kt * 32 + kh * 16;
          f0.u[0] = *(const v4u*)bp0;               // ds_load_b128 from persistent weights
          f0.u[1] = *(const v4u*)(bp0 + 8);
          const unsigned short* bp1 = bp0 + (size_t)16 * Ktot;
          f1.u[0] = *(const v4u*)bp1;
          f1.u[1] = *(const v4u*)(bp1 + 8);
          acc0 = __builtin_amdgcn_wmma_f32_16x16x32_bf16(false, a.bf, false, f0.bf, (short)0, acc0, false, false);
          acc1 = __builtin_amdgcn_wmma_f32_16x16x32_bf16(false, a.bf, false, f1.bf, (short)0, acc1, false, false);
        }
        wkoff += K;
      }
      // spill gate tiles to LDS: D layout = vgpr r -> M = mi*16 + kh*8 + r, lane%16 -> N
      {
        const int mbase = mi * 16 + kh * 8;
        #pragma unroll
        for (int r = 0; r < 8; ++r) {
          ldsG[(mbase + r) * 64 + nt0 * 16 + lm]       = acc0[r];
          ldsG[(mbase + r) * 64 + (nt0 + 1) * 16 + lm] = acc1[r];
        }
      }
      __syncthreads();
      // fused pointwise LSTM cell for this group's 16 hidden units x 64 batch
      for (int u = tid; u < BATCH * 16; u += 256) {
        int b = u >> 4, jl = u & 15;
        float gi = ldsG[b * 64 + jl]      + ldsB[jl];
        float gf = ldsG[b * 64 + 16 + jl] + ldsB[16 + jl];
        float gg = ldsG[b * 64 + 32 + jl] + ldsB[32 + jl];
        float go = ldsG[b * 64 + 48 + jl] + ldsB[48 + jl];
        int j = g * 16 + jl;
        size_t cIdx = (size_t)b * 2 * HIDDEN + (layer ? HIDDEN : 0) + j;
        float c  = outLC[cIdx];
        float cn = sigm(gf) * c + sigm(gi) * tanhf(gg);
        float hn = sigm(go) * tanhf(cn);
        outLC[cIdx] = cn;
        outLH[cIdx] = hn;
        unsigned short hb = f2bf(hn);
        if (layer == 0) {
          h0bf[(size_t)(p & 1) * BATCH * HIDDEN + (size_t)b * HIDDEN + j] = hb;
        } else {
          h1bf[(size_t)(s & 1) * BATCH * HIDDEN + (size_t)b * HIDDEN + j] = hb;
          outSeq[(size_t)b * SEQ * HIDDEN + (size_t)s * HIDDEN + j] = hn;
        }
      }
    }
    // layer0 groups prefetch/gather embeddings for step p+1 into the other parity buffer
    if (layer == 0 && (p + 1) < SEQ) {
      const int t = p + 1;
      for (int e = g * 256 + tid; e < BATCH * EMBED; e += 64 * 256) {
        int b = e / EMBED, col = e - b * EMBED;
        int tok = tokens[(size_t)b * SEQ + t];
        xbf[(size_t)(t & 1) * BATCH * EMBED + e] = f2bf(embed_w[(size_t)tok * EMBED + col]);
      }
    }
    // ---- device-scope barrier between phases ----
    __syncthreads();
    if (tid == 0) {
      __builtin_amdgcn_fence(__ATOMIC_RELEASE, "agent");
      unsigned e = __hip_atomic_load(&ctrl[1], __ATOMIC_RELAXED, __HIP_MEMORY_SCOPE_AGENT);
      unsigned a = __hip_atomic_fetch_add(&ctrl[0], 1u, __ATOMIC_RELAXED, __HIP_MEMORY_SCOPE_AGENT);
      if (a == (unsigned)(NBLOCKS - 1)) {
        __hip_atomic_store(&ctrl[0], 0u, __ATOMIC_RELAXED, __HIP_MEMORY_SCOPE_AGENT);
        __hip_atomic_fetch_add(&ctrl[1], 1u, __ATOMIC_RELEASE, __HIP_MEMORY_SCOPE_AGENT);
      } else {
        while (__hip_atomic_load(&ctrl[1], __ATOMIC_RELAXED, __HIP_MEMORY_SCOPE_AGENT) == e)
          __builtin_amdgcn_s_sleep(2);
      }
      __builtin_amdgcn_fence(__ATOMIC_ACQUIRE, "agent");
    }
    __syncthreads();
  }
}

extern "C" void kernel_launch(void* const* d_in, const int* in_sizes, int n_in,
                              void* d_out, int out_size, void* d_ws, size_t ws_size,
                              hipStream_t stream) {
  const int*   tokens  = (const int*)  d_in[0];
  const float* hidden  = (const float*)d_in[1];
  const float* cell    = (const float*)d_in[2];
  const float* embed_w = (const float*)d_in[3];
  const float* W_ih0   = (const float*)d_in[4];
  const float* W_hh0   = (const float*)d_in[5];
  const float* b0      = (const float*)d_in[6];
  const float* W_ih1   = (const float*)d_in[7];
  const float* W_hh1   = (const float*)d_in[8];
  const float* b1      = (const float*)d_in[9];
  float* out = (float*)d_out;
  unsigned char* ws = (unsigned char*)d_ws;

  (void)in_sizes; (void)n_in; (void)out_size; (void)ws_size;

  // allow 272KB of dynamic LDS (320KB/WGP on gfx1250)
  hipFuncSetAttribute((const void*)lstm_main,
                      hipFuncAttributeMaxDynamicSharedMemorySize, (int)SMEM_BYTES);

  lstm_init<<<64, 256, 0, stream>>>(tokens, hidden, cell, embed_w, out, ws);
  lstm_main<<<NBLOCKS, 256, SMEM_BYTES, stream>>>(tokens, embed_w,
                                                  W_ih0, W_hh0, b0,
                                                  W_ih1, W_hh1, b1,
                                                  out, ws);
}
